// GAT_15075335209498
// MI455X (gfx1250) — matrix operand order repaired and checked by reading, not compiled, not run on hardware
//
#include <hip/hip_runtime.h>
#include <stdint.h>

typedef __bf16 bf16_t;
typedef __attribute__((ext_vector_type(16))) __bf16 v16bf;
typedef __attribute__((ext_vector_type(8)))  __bf16 v8bf;
typedef __attribute__((ext_vector_type(8)))  float  v8f;

// ---------------- helpers ----------------
static __device__ __forceinline__ bf16_t f2bf(float f) {
  uint32_t u = __float_as_uint(f);
  uint32_t r = u + 0x7FFFu + ((u >> 16) & 1u);   // round-to-nearest-even
  union { uint16_t s; bf16_t b; } cv;
  cv.s = (uint16_t)(r >> 16);
  return cv.b;
}

static __device__ __forceinline__ v16bf ldfrag(const bf16_t* p0, const bf16_t* p1) {
  v8bf lo = *(const v8bf*)p0;
  v8bf hi = *(const v8bf*)p1;
  v16bf r;
#pragma unroll
  for (int i = 0; i < 8; ++i) { r[i] = lo[i]; r[i + 8] = hi[i]; }
  return r;
}

// ---------------- conversion kernels ----------------
__global__ void gat_f32_to_bf16(const float* __restrict__ s, bf16_t* __restrict__ d, size_t n) {
  size_t i = (size_t)blockIdx.x * blockDim.x + threadIdx.x;
  if (i < n) d[i] = f2bf(s[i]);
}

// Convert weight [Nsrc,K] f32 -> [Npad,K] bf16, zero-padding rows >= Nsrc.
__global__ void gat_w_pad_bf16(const float* __restrict__ W, bf16_t* __restrict__ d,
                               int Nsrc, int Npad, int K) {
  size_t i = (size_t)blockIdx.x * blockDim.x + threadIdx.x;
  size_t tot = (size_t)Npad * K;
  if (i >= tot) return;
  int row = (int)(i / K);
  int col = (int)(i % K);
  d[i] = (row < Nsrc) ? f2bf(W[(size_t)row * K + col]) : f2bf(0.f);
}

__global__ void gat_fill_u32(uint32_t* __restrict__ p, uint32_t v, size_t n) {
  size_t i = (size_t)blockIdx.x * blockDim.x + threadIdx.x;
  if (i < n) p[i] = v;
}

// ---------------- WMMA GEMM: C[M,N] = A[M,K] * B[N,K]^T ----------------
// A row-major bf16 (lda=K), B row-major bf16 (weights, ldb=K), C f32 (ldc=N).
// One wave computes a 16x64 tile: 4 accumulators, A fragment reused 4x.
// N must be a multiple of 64, K a multiple of 32.
__global__ void gat_gemm_bf16_nt(const bf16_t* __restrict__ A, const bf16_t* __restrict__ B,
                                 float* __restrict__ C, int M, int N, int K, int totalWaves) {
  int wid = (int)(blockIdx.x * (blockDim.x >> 5)) + (int)(threadIdx.x >> 5);
  if (wid >= totalWaves) return;           // wave-uniform: EXEC stays all-ones for WMMA
  int lane = threadIdx.x & 31;
  int hf   = lane >> 4;                    // 0: lanes 0-15, 1: lanes 16-31
  int l15  = lane & 15;
  int ntiles = N >> 6;
  int mt = wid / ntiles, nt = wid % ntiles;
  int m0 = mt << 4, n0 = nt << 6;

  int arow = m0 + l15; if (arow >= M) arow = M - 1;   // clamp (store guarded below)
  const bf16_t* Ap  = A + (size_t)arow * K;
  const bf16_t* Bp0 = B + (size_t)(n0 + l15) * K;
  const bf16_t* Bp1 = Bp0 + (size_t)16 * K;
  const bf16_t* Bp2 = Bp0 + (size_t)32 * K;
  const bf16_t* Bp3 = Bp0 + (size_t)48 * K;

  v8f acc0 = {}, acc1 = {}, acc2 = {}, acc3 = {};
  for (int k = 0; k < K; k += 32) {
    // A 16x32 fragment: lane<16 holds K k..k+15; lane>=16 holds K k+16..k+31
    v16bf a  = ldfrag(Ap + k + hf * 16, Ap + k + hf * 16 + 8);
    // B 32x16 fragment: lane<16 holds K k..k+15 of col n; lane>=16 holds K k+16..k+31
    int kb = k + hf * 16;
    v16bf b0 = ldfrag(Bp0 + kb, Bp0 + kb + 8);
    v16bf b1 = ldfrag(Bp1 + kb, Bp1 + kb + 8);
    v16bf b2 = ldfrag(Bp2 + kb, Bp2 + kb + 8);
    v16bf b3 = ldfrag(Bp3 + kb, Bp3 + kb + 8);
    acc0 = __builtin_amdgcn_wmma_f32_16x16x32_bf16(false, a, false, b0, (short)0, acc0, false, false);
    acc1 = __builtin_amdgcn_wmma_f32_16x16x32_bf16(false, a, false, b1, (short)0, acc1, false, false);
    acc2 = __builtin_amdgcn_wmma_f32_16x16x32_bf16(false, a, false, b2, (short)0, acc2, false, false);
    acc3 = __builtin_amdgcn_wmma_f32_16x16x32_bf16(false, a, false, b3, (short)0, acc3, false, false);
  }
  // C/D layout: VGPR i -> row m0 + hf*8 + i, col n0 + (lane&15)
#pragma unroll
  for (int i = 0; i < 8; ++i) {
    int row = m0 + hf * 8 + i;
    if (row < M) {
      float* Cr = C + (size_t)row * N + n0 + l15;
      Cr[0]  = acc0[i];
      Cr[16] = acc1[i];
      Cr[32] = acc2[i];
      Cr[48] = acc3[i];
    }
  }
}

// ---------------- attention dot: out[n,h] = sum_d F[n, h*od+d] * a[h*od+d] ----------------
__global__ void gat_row_dot(const float* __restrict__ F, const float* __restrict__ a,
                            float* __restrict__ out, int nNodes, int H, int od, int rs) {
  int wid  = (int)(((size_t)blockIdx.x * blockDim.x + threadIdx.x) >> 5);
  int lane = threadIdx.x & 31;
  if (wid >= nNodes * H) return;
  int n = wid / H, h = wid % H;
  const float* row = F + (size_t)n * rs + h * od;
  const float* av  = a + h * od;
  float sum = 0.f;
  for (int d = lane; d < od; d += 32) sum += row[d] * av[d];
#pragma unroll
  for (int o = 16; o >= 1; o >>= 1) sum += __shfl_down(sum, o, 32);
  if (lane == 0) out[wid] = sum;
}

// ---------------- edge pass 1: e = leaky_relu(el[src]+er[dst]); segment max via ordered-uint atomicMax
__global__ void gat_edge_max(const int* __restrict__ src, const int* __restrict__ dst,
                             const float* __restrict__ el, const float* __restrict__ er,
                             float* __restrict__ ebuf, uint32_t* __restrict__ menc, int E, int H) {
  int e = blockIdx.x * blockDim.x + threadIdx.x;
  if (e >= E) return;
  int s = src[e], d = dst[e];
  for (int h = 0; h < H; ++h) {
    float v = el[s * H + h] + er[d * H + h];
    v = (v > 0.f) ? v : 0.2f * v;
    ebuf[(size_t)e * H + h] = v;
    uint32_t bits = __float_as_uint(v);
    uint32_t enc  = (bits & 0x80000000u) ? ~bits : (bits | 0x80000000u);  // order-preserving
    atomicMax(&menc[d * H + h], enc);
  }
}

__global__ void gat_decode_m(const uint32_t* __restrict__ menc, float* __restrict__ m, size_t n) {
  size_t i = (size_t)blockIdx.x * blockDim.x + threadIdx.x;
  if (i >= n) return;
  uint32_t e = menc[i];
  uint32_t bits = (e & 0x80000000u) ? (e ^ 0x80000000u) : ~e;  // enc==0 -> NaN -> guarded to 0
  float f = __uint_as_float(bits);
  m[i] = __builtin_isfinite(f) ? f : 0.f;
}

// ---------------- edge pass 2: ex = exp(e - m[dst]); s[dst] += ex (ebuf overwritten in place)
__global__ void gat_edge_exp(const int* __restrict__ dst, float* __restrict__ ebuf,
                             const float* __restrict__ m, float* __restrict__ sden, int E, int H) {
  int e = blockIdx.x * blockDim.x + threadIdx.x;
  if (e >= E) return;
  int d = dst[e];
  for (int h = 0; h < H; ++h) {
    float ex = __expf(ebuf[(size_t)e * H + h] - m[d * H + h]);
    ebuf[(size_t)e * H + h] = ex;
    atomicAdd(&sden[d * H + h], ex);
  }
}

// ---------------- edge pass 3: agg[dst,:] += ex * fs[src,:]  (one wave per edge)
__global__ void gat_edge_agg(const int* __restrict__ src, const int* __restrict__ dst,
                             const float* __restrict__ ex, const float* __restrict__ fs,
                             float* __restrict__ agg, int E, int H, int od, int rs) {
  int wid  = (int)(((size_t)blockIdx.x * blockDim.x + threadIdx.x) >> 5);
  int lane = threadIdx.x & 31;
  if (wid >= E) return;
  int s = src[wid], d = dst[wid];
  int ft = H * od;
  const float* fr = fs + (size_t)s * rs;
  float* ag = agg + (size_t)d * rs;
  for (int c = lane; c < ft; c += 32) {
    float w = ex[(size_t)wid * H + c / od];
    atomicAdd(&ag[c], w * fr[c]);
  }
}

// ---------------- epilogues ----------------
__global__ void gat_epi_bf16(const float* __restrict__ agg, const float* __restrict__ sden,
                             const float* __restrict__ b, bf16_t* __restrict__ hout,
                             int Nd, int H, int od, int rs) {
  int ft = H * od;
  size_t tot = (size_t)Nd * ft;
  size_t i = (size_t)blockIdx.x * blockDim.x + threadIdx.x;
  if (i >= tot) return;
  int n = (int)(i / ft), c = (int)(i % ft);
  float den = fmaxf(sden[n * H + c / od], 1e-9f);
  float v = agg[(size_t)n * rs + c] / den + b[c];
  v = (v > 0.f) ? v : expm1f(v);                     // ELU
  hout[i] = f2bf(v);
}

__global__ void gat_epi_f32(float* __restrict__ agg, const float* __restrict__ sden,
                            const float* __restrict__ b, int Nd, int H, int od, int rs) {
  int ft = H * od;
  size_t tot = (size_t)Nd * ft;
  size_t i = (size_t)blockIdx.x * blockDim.x + threadIdx.x;
  if (i >= tot) return;
  int n = (int)(i / ft), c = (int)(i % ft);
  float den = fmaxf(sden[n * H + c / od], 1e-9f);
  agg[(size_t)n * rs + c] = agg[(size_t)n * rs + c] / den + b[c];   // no activation
}

// ---------------- head-mean + log-softmax (one wave per node) ----------------
__global__ void gat_mean_lsm(const float* __restrict__ rst, float* __restrict__ out,
                             int Nd, int H, int C, int rs) {
  int wid  = (int)(((size_t)blockIdx.x * blockDim.x + threadIdx.x) >> 5);
  int lane = threadIdx.x & 31;
  if (wid >= Nd) return;
  const float* row = rst + (size_t)wid * rs;
  float v0 = -INFINITY, v1 = -INFINITY;
  int c0 = lane, c1 = lane + 32;
  if (c0 < C) { float a = 0.f; for (int h = 0; h < H; ++h) a += row[h * C + c0]; v0 = a / (float)H; }
  if (c1 < C) { float a = 0.f; for (int h = 0; h < H; ++h) a += row[h * C + c1]; v1 = a / (float)H; }
  float mx = fmaxf(v0, v1);
#pragma unroll
  for (int o = 16; o >= 1; o >>= 1) mx = fmaxf(mx, __shfl_xor(mx, o, 32));
  float se = (c0 < C ? __expf(v0 - mx) : 0.f) + (c1 < C ? __expf(v1 - mx) : 0.f);
#pragma unroll
  for (int o = 16; o >= 1; o >>= 1) se += __shfl_xor(se, o, 32);
  float ls = __logf(se);
  if (c0 < C) out[(size_t)wid * C + c0] = v0 - mx - ls;
  if (c1 < C) out[(size_t)wid * C + c1] = v1 - mx - ls;
}

// ---------------- host orchestration ----------------
extern "C" void kernel_launch(void* const* d_in, const int* in_sizes, int n_in,
                              void* d_out, int out_size, void* d_ws, size_t ws_size,
                              hipStream_t stream) {
  (void)in_sizes; (void)n_in; (void)out_size; (void)ws_size;

  const float* x = (const float*)d_in[0];
  const int* srcs[3] = {(const int*)d_in[1], (const int*)d_in[3], (const int*)d_in[5]};
  const int* dsts[3] = {(const int*)d_in[2], (const int*)d_in[4], (const int*)d_in[6]};
  const float *Wsrc[3], *Wdst[3], *al[3], *ar[3], *bias[3];
  for (int l = 0; l < 3; ++l) {
    Wsrc[l] = (const float*)d_in[7 + 5 * l + 0];
    Wdst[l] = (const float*)d_in[7 + 5 * l + 1];
    al[l]   = (const float*)d_in[7 + 5 * l + 2];
    ar[l]   = (const float*)d_in[7 + 5 * l + 3];
    bias[l] = (const float*)d_in[7 + 5 * l + 4];
  }

  const int H = 4;
  const int NS[3]   = {150000, 25000, 6000};
  const int ND[3]   = {25000, 6000, 1024};
  const int NE[3]   = {375000, 90000, 10240};
  const int KIN[3]  = {256, 512, 512};
  const int OUTD[3] = {128, 128, 47};
  const int NPAD[3] = {512, 512, 192};   // N padded to multiple of 64

  char* base = (char*)d_ws;
  size_t off = 0;
  auto alloc = [&](size_t bytes) -> void* {
    void* p = base + off;
    off = (off + bytes + 255) & ~(size_t)255;
    return p;
  };

  // persistent allocations
  bf16_t* wS[3]; bf16_t* wD[3];
  for (int l = 0; l < 3; ++l) {
    wS[l] = (bf16_t*)alloc((size_t)NPAD[l] * KIN[l] * 2);
    wD[l] = (bf16_t*)alloc((size_t)NPAD[l] * KIN[l] * 2);
  }
  bf16_t* h1bf = (bf16_t*)alloc((size_t)ND[0] * 512 * 2);
  bf16_t* h2bf = (bf16_t*)alloc((size_t)ND[1] * 512 * 2);
  const size_t scratch0 = off;

  // convert + pad weights (L2-resident thereafter)
  for (int l = 0; l < 3; ++l) {
    int nsrc = H * OUTD[l];
    size_t tot = (size_t)NPAD[l] * KIN[l];
    int blks = (int)((tot + 255) / 256);
    gat_w_pad_bf16<<<blks, 256, 0, stream>>>(Wsrc[l], wS[l], nsrc, NPAD[l], KIN[l]);
    gat_w_pad_bf16<<<blks, 256, 0, stream>>>(Wdst[l], wD[l], nsrc, NPAD[l], KIN[l]);
  }

  float* finalRst = nullptr;
  for (int l = 0; l < 3; ++l) {
    off = scratch0;                           // reuse scratch region each layer
    int Ns = NS[l], Nd = ND[l], E = NE[l], K = KIN[l], od = OUTD[l], Np = NPAD[l];

    bf16_t* A;
    if (l == 0) {
      bf16_t* xbf = (bf16_t*)alloc((size_t)Ns * K * 2);
      size_t tot = (size_t)Ns * K;
      gat_f32_to_bf16<<<(int)((tot + 255) / 256), 256, 0, stream>>>(x, xbf, tot);
      A = xbf;
    } else {
      A = (l == 1) ? h1bf : h2bf;
    }

    float*    fs   = (float*)alloc((size_t)Ns * Np * 4);
    float*    fd   = (float*)alloc((size_t)Nd * Np * 4);
    float*    el   = (float*)alloc((size_t)Ns * H * 4);
    float*    er   = (float*)alloc((size_t)Nd * H * 4);
    float*    ebuf = (float*)alloc((size_t)E * H * 4);
    uint32_t* menc = (uint32_t*)alloc((size_t)Nd * H * 4);
    float*    m    = (float*)alloc((size_t)Nd * H * 4);
    float*    sden = (float*)alloc((size_t)Nd * H * 4);
    float*    agg  = (float*)alloc((size_t)Nd * Np * 4);

    // GEMMs (WMMA bf16)
    {
      int mt = (Ns + 15) / 16, tw = mt * (Np / 64);
      gat_gemm_bf16_nt<<<(tw + 7) / 8, 256, 0, stream>>>(A, wS[l], fs, Ns, Np, K, tw);
      int mtd = (Nd + 15) / 16, twd = mtd * (Np / 64);
      gat_gemm_bf16_nt<<<(twd + 7) / 8, 256, 0, stream>>>(A, wD[l], fd, Nd, Np, K, twd);
    }
    // attention dots
    {
      int w1 = Ns * H;
      gat_row_dot<<<(int)(((size_t)w1 * 32 + 255) / 256), 256, 0, stream>>>(fs, al[l], el, Ns, H, od, Np);
      int w2 = Nd * H;
      gat_row_dot<<<(int)(((size_t)w2 * 32 + 255) / 256), 256, 0, stream>>>(fd, ar[l], er, Nd, H, od, Np);
    }
    // zero-init reductions
    size_t nmh = (size_t)Nd * H;
    gat_fill_u32<<<(int)((nmh + 255) / 256), 256, 0, stream>>>(menc, 0u, nmh);
    gat_fill_u32<<<(int)((nmh + 255) / 256), 256, 0, stream>>>((uint32_t*)sden, 0u, nmh);
    size_t nagg = (size_t)Nd * Np;
    gat_fill_u32<<<(int)((nagg + 255) / 256), 256, 0, stream>>>((uint32_t*)agg, 0u, nagg);
    // edge softmax + aggregation
    gat_edge_max<<<(E + 255) / 256, 256, 0, stream>>>(srcs[l], dsts[l], el, er, ebuf, menc, E, H);
    gat_decode_m<<<(int)((nmh + 255) / 256), 256, 0, stream>>>(menc, m, nmh);
    gat_edge_exp<<<(E + 255) / 256, 256, 0, stream>>>(dsts[l], ebuf, m, sden, E, H);
    gat_edge_agg<<<(int)(((size_t)E * 32 + 255) / 256), 256, 0, stream>>>(srcs[l], dsts[l], ebuf, fs, agg, E, H, od, Np);
    // epilogue
    size_t nout = (size_t)Nd * (H * od);
    if (l < 2) {
      bf16_t* hout = (l == 0) ? h1bf : h2bf;
      gat_epi_bf16<<<(int)((nout + 255) / 256), 256, 0, stream>>>(agg, sden, bias[l], hout, Nd, H, od, Np);
    } else {
      gat_epi_f32<<<(int)((nout + 255) / 256), 256, 0, stream>>>(agg, sden, bias[l], Nd, H, od, Np);
      finalRst = agg;
    }
  }

  gat_mean_lsm<<<(int)(((size_t)ND[2] * 32 + 255) / 256), 256, 0, stream>>>(
      finalRst, (float*)d_out, ND[2], H, 47, NPAD[2]);
}
